// DiTBlock_88115549045156
// MI455X (gfx1250) — compile-verified, matrix operand order checked by reading, not run on hardware
//
#include <hip/hip_runtime.h>

#define H    1024
#define NHD  16
#define HDIM 64
#define CDIM 1024
#define MLPD 4096
#define LSEQ 2048
#define BATCH 2
#define MTOT (BATCH*LSEQ)
#define EPSLN 1e-5f

typedef __attribute__((ext_vector_type(16))) _Float16 v16h;
typedef __attribute__((ext_vector_type(8)))  float    v8f;

// ---------------------------------------------------------------------------
// Optional CDNA5 async global->LDS path (ASYNCcnt). Guarded so the file
// still compiles if this toolchain does not declare the builtins.
// Probe-derived prototype: (v4i AS1* src, v4i AS3* lds, imm off, imm cpol).
// ---------------------------------------------------------------------------
#if __has_builtin(__builtin_amdgcn_global_load_async_to_lds_b128) && \
    __has_builtin(__builtin_amdgcn_s_wait_asynccnt)
#define USE_ASYNC_LDS 1
typedef int v4i_ __attribute__((vector_size(16)));
typedef __attribute__((address_space(1))) v4i_ as1_v4i;
typedef __attribute__((address_space(3))) v4i_ as3_v4i;
__device__ inline void async_cp16(const _Float16* gsrc, _Float16* ldst) {
    // LDS generic address truncates to the LDS offset (aperture rule).
    __builtin_amdgcn_global_load_async_to_lds_b128(
        (as1_v4i*)(uintptr_t)gsrc,
        (as3_v4i*)(unsigned)(uintptr_t)ldst, 0, 0);
}
__device__ inline void async_wait_all() { __builtin_amdgcn_s_wait_asynccnt(0); }
#else
__device__ inline void async_wait_all() {}
#endif

// ---------------------------------------------------------------------------
// WMMA helpers (CDNA5 16x16x32 f16 -> f32 accumulate)
// ---------------------------------------------------------------------------
__device__ inline v8f wmma16(v16h a, v16h b, v8f c) {
    return __builtin_amdgcn_wmma_f32_16x16x32_f16(
        /*neg_a=*/false, a, /*neg_b=*/false, b,
        /*c_mod=*/(short)0, c, /*reuse_a=*/false, /*reuse_b=*/false);
}

// A/B fragment from a row-major 16x32 tile (leading dim `ld` halves).
// ISA 7.12.2 16-bit layout: lane&15 = row (A) / col (B with row-major W),
// VGPR0..3 hold K pairs {0..7}(lane<16)/{8..15}(lane>=16), VGPR4..7 hold +16.
__device__ inline v16h frag_load_row(const _Float16* base, int ld, int lane) {
    const int row = lane & 15;
    const int hi  = (lane & 16) ? 8 : 0;
    const _Float16* p = base + (size_t)row * ld;
    union { uint4 u[2]; v16h h; } u;
    u.u[0] = *(const uint4*)(p + hi);        // K = hi .. hi+7
    u.u[1] = *(const uint4*)(p + 16 + hi);   // K = 16+hi .. 23+hi
    return u.h;
}

// ---------------------------------------------------------------------------
// fp32 -> fp16 conversion
// ---------------------------------------------------------------------------
__global__ __launch_bounds__(256) void cvt_kernel(const float* __restrict__ in,
                                                  _Float16* __restrict__ out, int n) {
    int i = blockIdx.x * 256 + threadIdx.x;
    if (i < n) out[i] = (_Float16)in[i];
}

// ---------------------------------------------------------------------------
// ss = cond @ W.T + b   (B=2 rows only -> scalar dot per output)
// ---------------------------------------------------------------------------
__global__ __launch_bounds__(256) void cond_gemm_kernel(const float* __restrict__ cond,
                                                        const float* __restrict__ W,
                                                        const float* __restrict__ bias,
                                                        float* __restrict__ ss) {
    int idx = blockIdx.x * 256 + threadIdx.x;         // over BATCH * 2H
    if (idx >= BATCH * 2 * H) return;
    int b = idx >> 11;                                // / 2048
    int n = idx & 2047;
    const float* c = cond + (size_t)b * CDIM;
    const float* w = W + (size_t)n * CDIM;
    float acc = bias[n];
    for (int kk = 0; kk < CDIM; ++kk) acc += c[kk] * w[kk];
    ss[idx] = acc;
}

// ---------------------------------------------------------------------------
// AdaLN: y = LN(x) * (1 + scale) + shift, f16 out.  One block per row.
// ---------------------------------------------------------------------------
__global__ __launch_bounds__(256) void adaln_kernel(const float* __restrict__ x,
                                                    const float* __restrict__ ss,
                                                    _Float16* __restrict__ out) {
    const int row = blockIdx.x;          // [0, B*L)
    const int b   = row >> 11;           // row / 2048
    const float* xr = x + (size_t)row * H;
    float vbuf[4];
    float s = 0.f, s2 = 0.f;
#pragma unroll
    for (int i = 0; i < 4; ++i) {
        float u = xr[threadIdx.x + i * 256];
        vbuf[i] = u; s += u; s2 += u * u;
    }
#pragma unroll
    for (int off = 16; off > 0; off >>= 1) {
        s  += __shfl_xor(s,  off, 32);
        s2 += __shfl_xor(s2, off, 32);
    }
    __shared__ float rs_[8], rs2_[8], stats[2];
    const int wave = threadIdx.x >> 5;
    if ((threadIdx.x & 31) == 0) { rs_[wave] = s; rs2_[wave] = s2; }
    __syncthreads();
    if (threadIdx.x == 0) {
        float ts = 0.f, ts2 = 0.f;
        for (int i = 0; i < 8; ++i) { ts += rs_[i]; ts2 += rs2_[i]; }
        float mu  = ts * (1.0f / H);
        float var = ts2 * (1.0f / H) - mu * mu;
        stats[0] = mu;
        stats[1] = rsqrtf(var + EPSLN);
    }
    __syncthreads();
    const float mu = stats[0], rsq = stats[1];
    const float* ssb = ss + (size_t)b * 2 * H;
#pragma unroll
    for (int i = 0; i < 4; ++i) {
        int col = threadIdx.x + i * 256;
        float y  = (vbuf[i] - mu) * rsq;
        float sc = ssb[col];
        float sh = ssb[H + col];
        out[(size_t)row * H + col] = (_Float16)(y * (1.f + sc) + sh);
    }
}

// ---------------------------------------------------------------------------
// Generic WMMA GEMM: C[M,N] = A[M,K] @ W[N,K]^T + bias
//   mode 0: store f16                 mode 1: store f32 + residual add
//   mode 2: exact GELU -> f16         mode 3: store f16 TRANSPOSED (out[N][M])
// Block tile 128x128, K-step 32, double-buffered LDS, 8 waves, each wave
// a 32x64 tile = 2x4 WMMA subtiles (8 v_wmma per K-step, 12 ds_load_b128).
// ---------------------------------------------------------------------------
__device__ inline void stage_tile(const _Float16* __restrict__ A, int lda, int m0,
                                  const _Float16* __restrict__ W, int ldw, int n0,
                                  int k0, _Float16* As, _Float16* Bs, int t) {
    const int r0 = t >> 2,          c0 = (t & 3) * 8;
    const int r1 = (t + 256) >> 2,  c1 = c0;
#if defined(USE_ASYNC_LDS)
    async_cp16(&A[(size_t)(m0 + r0) * lda + k0 + c0], &As[r0 * 40 + c0]);
    async_cp16(&A[(size_t)(m0 + r1) * lda + k0 + c1], &As[r1 * 40 + c1]);
    async_cp16(&W[(size_t)(n0 + r0) * ldw + k0 + c0], &Bs[r0 * 40 + c0]);
    async_cp16(&W[(size_t)(n0 + r1) * ldw + k0 + c1], &Bs[r1 * 40 + c1]);
#else
    *(uint4*)&As[r0 * 40 + c0] = *(const uint4*)&A[(size_t)(m0 + r0) * lda + k0 + c0];
    *(uint4*)&As[r1 * 40 + c1] = *(const uint4*)&A[(size_t)(m0 + r1) * lda + k0 + c1];
    *(uint4*)&Bs[r0 * 40 + c0] = *(const uint4*)&W[(size_t)(n0 + r0) * ldw + k0 + c0];
    *(uint4*)&Bs[r1 * 40 + c1] = *(const uint4*)&W[(size_t)(n0 + r1) * ldw + k0 + c1];
#endif
}

__global__ __launch_bounds__(256) void gemm_wmma(
    const _Float16* __restrict__ A, int lda,
    const _Float16* __restrict__ W, int ldw,
    const float* __restrict__ bias,
    int M, int N, int K, int mode,
    const float* __restrict__ residual,
    float* __restrict__ outF, _Float16* __restrict__ outH) {
    __shared__ _Float16 As[2][128 * 40];   // padded ld = 40 halves (80B rows)
    __shared__ _Float16 Bs[2][128 * 40];
    const int t    = threadIdx.x;
    const int lane = t & 31;
    const int wave = t >> 5;
    const int wm = (wave & 3) * 32;        // 4 waves tile M (128)
    const int wn = (wave >> 2) * 64;       // 2 waves tile N (128)
    const int m0 = blockIdx.x * 128;
    const int n0 = blockIdx.y * 128;

    v8f acc[2][4] = {{{}, {}, {}, {}}, {{}, {}, {}, {}}};

    int cur = 0;
    stage_tile(A, lda, m0, W, ldw, n0, 0, As[0], Bs[0], t);
    async_wait_all();
    __syncthreads();

    for (int k0 = 0; k0 < K; k0 += 32) {
        const int nxt = cur ^ 1;
        if (k0 + 32 < K)
            stage_tile(A, lda, m0, W, ldw, n0, k0 + 32, As[nxt], Bs[nxt], t);

        v16h af0 = frag_load_row(&As[cur][(wm)      * 40], 40, lane);
        v16h af1 = frag_load_row(&As[cur][(wm + 16) * 40], 40, lane);
        v16h bf0 = frag_load_row(&Bs[cur][(wn)      * 40], 40, lane);
        v16h bf1 = frag_load_row(&Bs[cur][(wn + 16) * 40], 40, lane);
        v16h bf2 = frag_load_row(&Bs[cur][(wn + 32) * 40], 40, lane);
        v16h bf3 = frag_load_row(&Bs[cur][(wn + 48) * 40], 40, lane);
        acc[0][0] = wmma16(af0, bf0, acc[0][0]);
        acc[0][1] = wmma16(af0, bf1, acc[0][1]);
        acc[0][2] = wmma16(af0, bf2, acc[0][2]);
        acc[0][3] = wmma16(af0, bf3, acc[0][3]);
        acc[1][0] = wmma16(af1, bf0, acc[1][0]);
        acc[1][1] = wmma16(af1, bf1, acc[1][1]);
        acc[1][2] = wmma16(af1, bf2, acc[1][2]);
        acc[1][3] = wmma16(af1, bf3, acc[1][3]);

        async_wait_all();
        __syncthreads();
        cur = nxt;
    }

    const int rr = (lane >> 4) * 8;        // C layout: row = r + 8*(lane>=16)
    const int cc = lane & 15;              // col = lane&15
    if (mode == 3) {
        // Transposed f16 store: lane owns a fixed col and 8 consecutive rows
        // -> one b128 store per subtile.  out[N][M].
#pragma unroll
        for (int sm = 0; sm < 2; ++sm)
#pragma unroll
        for (int sn = 0; sn < 4; ++sn) {
            const int col  = n0 + wn + sn * 16 + cc;
            const int row0 = m0 + wm + sm * 16 + rr;
            union { _Float16 h[8]; uint4 u; } pk;
            const float bv = bias[col];
#pragma unroll
            for (int r = 0; r < 8; ++r) pk.h[r] = (_Float16)(acc[sm][sn][r] + bv);
            *(uint4*)&outH[(size_t)col * M + row0] = pk.u;
        }
        return;
    }
#pragma unroll
    for (int sm = 0; sm < 2; ++sm)
#pragma unroll
    for (int sn = 0; sn < 4; ++sn)
#pragma unroll
    for (int r = 0; r < 8; ++r) {
        const int row = m0 + wm + sm * 16 + rr + r;
        const int col = n0 + wn + sn * 16 + cc;
        float val = acc[sm][sn][r] + bias[col];
        const size_t idx = (size_t)row * N + col;
        if (mode == 0) {
            outH[idx] = (_Float16)val;
        } else if (mode == 1) {
            outF[idx] = val + residual[idx];
        } else {
            outH[idx] = (_Float16)(0.5f * val * (1.0f + erff(val * 0.70710678118654752f)));
        }
    }
}

// ---------------------------------------------------------------------------
// Flash attention: 4 independent waves per block, 1 wave per 16-query tile.
// S = Q K^T (4 WMMAs / 32-key chunk), online softmax, P V (4 WMMAs) with V
// pre-transposed (vT[H][MTOT]) so B-fragments are two b128 loads each.
// ---------------------------------------------------------------------------
__global__ __launch_bounds__(128) void attn_kernel(
    const _Float16* __restrict__ q,
    const _Float16* __restrict__ k,
    const _Float16* __restrict__ vT,
    const unsigned char* __restrict__ mask,
    _Float16* __restrict__ out) {
    __shared__ _Float16 Pl[4][16 * 40];
    const int lane = threadIdx.x & 31;
    const int wv   = threadIdx.x >> 5;
    const int qt = blockIdx.x * 4 + wv;        // query tile
    const int h  = blockIdx.y;                 // head
    const int b  = blockIdx.z;                 // batch
    const int q0 = b * LSEQ + qt * 16;         // global row
    const size_t hoff = (size_t)h * HDIM;

    const v16h qa0 = frag_load_row(q + (size_t)q0 * H + hoff,      H, lane);
    const v16h qa1 = frag_load_row(q + (size_t)q0 * H + hoff + 32, H, lane);

    v8f o0 = {}, o1 = {}, o2 = {}, o3 = {};
    float mrun[8], lrun[8];
#pragma unroll
    for (int r = 0; r < 8; ++r) { mrun[r] = -1e30f; lrun[r] = 0.f; }

    for (int kc = 0; kc < LSEQ; kc += 32) {
        v8f s0 = {}, s1 = {};
        {
            const _Float16* kb = k + (size_t)(b * LSEQ + kc) * H + hoff;
            s0 = wmma16(qa0, frag_load_row(kb,                      H, lane), s0);
            s0 = wmma16(qa1, frag_load_row(kb + 32,                 H, lane), s0);
            s1 = wmma16(qa0, frag_load_row(kb + (size_t)16 * H,      H, lane), s1);
            s1 = wmma16(qa1, frag_load_row(kb + (size_t)16 * H + 32, H, lane), s1);
        }
        // scale (1/sqrt(64)) + key padding mask
        const int key0 = kc + (lane & 15);
        const bool m0b = mask[b * LSEQ + key0]      != 0;
        const bool m1b = mask[b * LSEQ + key0 + 16] != 0;
#pragma unroll
        for (int r = 0; r < 8; ++r) {
            s0[r] = m0b ? -1e30f : s0[r] * 0.125f;
            s1[r] = m1b ? -1e30f : s1[r] * 0.125f;
        }
        // online softmax: rows live in 16-lane halves of the wave
        float alpha[8];
#pragma unroll
        for (int r = 0; r < 8; ++r) {
            float mx = fmaxf(s0[r], s1[r]);
#pragma unroll
            for (int off = 1; off < 16; off <<= 1)
                mx = fmaxf(mx, __shfl_xor(mx, off, 16));
            float mnew = fmaxf(mrun[r], mx);
            alpha[r] = __expf(mrun[r] - mnew);
            mrun[r] = mnew;
            s0[r] = __expf(s0[r] - mnew);
            s1[r] = __expf(s1[r] - mnew);
            float rs = s0[r] + s1[r];
#pragma unroll
            for (int off = 1; off < 16; off <<= 1)
                rs += __shfl_xor(rs, off, 16);
            lrun[r] = lrun[r] * alpha[r] + rs;
        }
#pragma unroll
        for (int r = 0; r < 8; ++r) {
            o0[r] *= alpha[r]; o1[r] *= alpha[r];
            o2[r] *= alpha[r]; o3[r] *= alpha[r];
        }
        // C-layout P -> row-major f16 tile in LDS -> A fragment
        __syncthreads();
        {
            const int rr = (lane >> 4) * 8;
            const int cc = lane & 15;
#pragma unroll
            for (int r = 0; r < 8; ++r) {
                Pl[wv][(rr + r) * 40 + cc]      = (_Float16)s0[r];
                Pl[wv][(rr + r) * 40 + cc + 16] = (_Float16)s1[r];
            }
        }
        __syncthreads();
        const v16h pa = frag_load_row(Pl[wv], 40, lane);
        // V^T: col (output dim) fixed per lane, key index contiguous.
        const _Float16* vtb = vT + hoff * MTOT + (size_t)b * LSEQ + kc;
        o0 = wmma16(pa, frag_load_row(vtb,                       MTOT, lane), o0);
        o1 = wmma16(pa, frag_load_row(vtb + (size_t)16 * MTOT,   MTOT, lane), o1);
        o2 = wmma16(pa, frag_load_row(vtb + (size_t)32 * MTOT,   MTOT, lane), o2);
        o3 = wmma16(pa, frag_load_row(vtb + (size_t)48 * MTOT,   MTOT, lane), o3);
    }
    const int rr = (lane >> 4) * 8;
    const int cc = lane & 15;
#pragma unroll
    for (int r = 0; r < 8; ++r) {
        const float inv = 1.f / lrun[r];
        const size_t ro = (size_t)(q0 + rr + r) * H + hoff;
        out[ro + cc]      = (_Float16)(o0[r] * inv);
        out[ro + 16 + cc] = (_Float16)(o1[r] * inv);
        out[ro + 32 + cc] = (_Float16)(o2[r] * inv);
        out[ro + 48 + cc] = (_Float16)(o3[r] * inv);
    }
}

// ---------------------------------------------------------------------------
// Host-side orchestration
// ---------------------------------------------------------------------------
extern "C" void kernel_launch(void* const* d_in, const int* in_sizes, int n_in,
                              void* d_out, int out_size, void* d_ws, size_t ws_size,
                              hipStream_t stream) {
    (void)in_sizes; (void)n_in; (void)out_size; (void)ws_size;
    const float* x     = (const float*)d_in[0];
    const float* cond  = (const float*)d_in[1];
    const unsigned char* mask = (const unsigned char*)d_in[2];
    const float* w_ad1 = (const float*)d_in[3];
    const float* b_ad1 = (const float*)d_in[4];
    const float* w_ad2 = (const float*)d_in[5];
    const float* b_ad2 = (const float*)d_in[6];
    const float* wq = (const float*)d_in[7];   const float* bq = (const float*)d_in[8];
    const float* wk = (const float*)d_in[9];   const float* bk = (const float*)d_in[10];
    const float* wv = (const float*)d_in[11];  const float* bv = (const float*)d_in[12];
    const float* wo = (const float*)d_in[13];  const float* bo = (const float*)d_in[14];
    const float* w1 = (const float*)d_in[15];  const float* b1 = (const float*)d_in[16];
    const float* w2 = (const float*)d_in[17];  const float* b2 = (const float*)d_in[18];

    char* p = (char*)d_ws;
    auto bump = [&](size_t bytes) -> char* {
        char* r = p;
        p += (bytes + 255) & ~(size_t)255;
        return r;
    };
    _Float16* wq16 = (_Float16*)bump((size_t)H * H * 2);
    _Float16* wk16 = (_Float16*)bump((size_t)H * H * 2);
    _Float16* wv16 = (_Float16*)bump((size_t)H * H * 2);
    _Float16* wo16 = (_Float16*)bump((size_t)H * H * 2);
    _Float16* w116 = (_Float16*)bump((size_t)MLPD * H * 2);
    _Float16* w216 = (_Float16*)bump((size_t)H * MLPD * 2);
    float*    ss1  = (float*)bump((size_t)BATCH * 2 * H * 4);
    float*    ss2  = (float*)bump((size_t)BATCH * 2 * H * 4);
    _Float16* nrm  = (_Float16*)bump((size_t)MTOT * H * 2);
    _Float16* qb   = (_Float16*)bump((size_t)MTOT * H * 2);
    _Float16* kb   = (_Float16*)bump((size_t)MTOT * H * 2);
    _Float16* vT   = (_Float16*)bump((size_t)H * MTOT * 2);   // transposed V
    _Float16* ab   = (_Float16*)bump((size_t)MTOT * H * 2);
    float*    x2   = (float*)bump((size_t)MTOT * H * 4);
    _Float16* hb   = (_Float16*)bump((size_t)MTOT * MLPD * 2);

    auto cvt = [&](const float* src, _Float16* dst, int n) {
        cvt_kernel<<<dim3((n + 255) / 256), dim3(256), 0, stream>>>(src, dst, n);
    };
    cvt(wq, wq16, H * H);
    cvt(wk, wk16, H * H);
    cvt(wv, wv16, H * H);
    cvt(wo, wo16, H * H);
    cvt(w1, w116, MLPD * H);
    cvt(w2, w216, H * MLPD);

    cond_gemm_kernel<<<dim3((BATCH * 2 * H + 255) / 256), dim3(256), 0, stream>>>(cond, w_ad1, b_ad1, ss1);
    cond_gemm_kernel<<<dim3((BATCH * 2 * H + 255) / 256), dim3(256), 0, stream>>>(cond, w_ad2, b_ad2, ss2);

    adaln_kernel<<<dim3(MTOT), dim3(256), 0, stream>>>(x, ss1, nrm);

    const dim3 gH(MTOT / 128, H / 128);
    gemm_wmma<<<gH, dim3(256), 0, stream>>>(nrm, H, wq16, H, bq, MTOT, H, H, 0, nullptr, nullptr, qb);
    gemm_wmma<<<gH, dim3(256), 0, stream>>>(nrm, H, wk16, H, bk, MTOT, H, H, 0, nullptr, nullptr, kb);
    gemm_wmma<<<gH, dim3(256), 0, stream>>>(nrm, H, wv16, H, bv, MTOT, H, H, 3, nullptr, nullptr, vT);

    attn_kernel<<<dim3(LSEQ / 64, NHD, BATCH), dim3(128), 0, stream>>>(qb, kb, vT, mask, ab);

    // x2 = x + attn_out @ wo.T + bo
    gemm_wmma<<<gH, dim3(256), 0, stream>>>(ab, H, wo16, H, bo, MTOT, H, H, 1, x, x2, nullptr);

    adaln_kernel<<<dim3(MTOT), dim3(256), 0, stream>>>(x2, ss2, nrm);

    // h = gelu(normed @ w1.T + b1)
    const dim3 gM(MTOT / 128, MLPD / 128);
    gemm_wmma<<<gM, dim3(256), 0, stream>>>(nrm, H, w116, H, b1, MTOT, MLPD, H, 2, nullptr, nullptr, hb);

    // out = x2 + h @ w2.T + b2
    gemm_wmma<<<gH, dim3(256), 0, stream>>>(hb, MLPD, w216, MLPD, b2, MTOT, H, MLPD, 1, x2, (float*)d_out, nullptr);
}
